// VJEPA2JointDiffuserRopeAttention_1176821039787
// MI455X (gfx1250) — matrix operand order repaired and checked by reading, not compiled, not run on hardware
//
#include <hip/hip_runtime.h>

// ---------------- types ----------------
typedef __attribute__((ext_vector_type(16))) __bf16        bf16x16;
typedef __attribute__((ext_vector_type(8)))  float         f32x8;
typedef __attribute__((ext_vector_type(4)))  unsigned int  u32x4;
typedef __attribute__((ext_vector_type(8)))  int           i32x8;
typedef __attribute__((ext_vector_type(4)))  int           i32x4;

union Frag16 { bf16x16 v; u32x4 q[2]; };

__device__ __forceinline__ unsigned short f2bf(float f) {
  unsigned int u = __float_as_uint(f);
  u += 0x7FFFu + ((u >> 16) & 1u);          // round-to-nearest-even
  return (unsigned short)(u >> 16);
}

// D = A*B + C, bf16 inputs, f32 accum
#define WMMA_BF16(C, A, B) \
  __builtin_amdgcn_wmma_f32_16x16x32_bf16(false, (A), false, (B), (short)0, (C), false, false)

// ---------------- problem constants ----------------
constexpr int HID   = 1024;
constexpr int HEADS = 16;
constexpr int HD    = 64;
constexpr int NV    = 2048;   // video tokens per batch
constexpr int NA    = 16;     // action tokens per batch
constexpr int NT    = NV + NA;        // 2064
constexpr int NP    = 2112;           // padded tokens (multiple of 64)
constexpr int BZ    = 2;

// ---------------- TDM: 2-D tile load global -> LDS (D# per CDNA5 ISA) ----
// Fixed tile shape 32 (K) x 128 (rows); LDS rows get 32 bf16 + 16-element pad ->
// 48-element row stride. pad_interval code 3 = 16 DWORDs, pad_amount code 7 = 8 DWORDs.
__device__ __forceinline__ void tdm_load_tile(unsigned lds_byte_addr, const void* gptr,
                                              unsigned tensor_d1 /* rows from tile */) {
  constexpr unsigned K = (unsigned)HID;
  unsigned long long ga = (unsigned long long)gptr;
  u32x4 g0;
  g0[0] = 1u;                                             // count=1, user descriptor
  g0[1] = lds_byte_addr;                                  // D#.lds_addr
  g0[2] = (unsigned)(ga & 0xFFFFFFFFu);                   // global_addr[31:0]
  g0[3] = (unsigned)((ga >> 32) & 0x01FFFFFFu)            // global_addr[56:32]
        | (2u << 30);                                     // type = 2 ("image")
  i32x8 g1;
  g1[0] = (int)((1u << 16)        // data_size = 1 -> 2 bytes
              | (1u << 20)        // pad_enable
              | (3u << 22)        // pad_interval: 16 DWORDs
              | (7u << 25));      // pad_amount:   8 DWORDs
  g1[1] = (int)((K & 0xFFFFu) << 16);                                 // tensor_d0 lo
  g1[2] = (int)(((K >> 16) & 0xFFFFu) | ((tensor_d1 & 0xFFFFu) << 16));
  g1[3] = (int)(((tensor_d1 >> 16) & 0xFFFFu) | (32u << 16));         // tile_d0 = 32
  g1[4] = (int)128u;                                                  // tile_d1 = 128
  g1[5] = (int)K;                                                     // td0_stride lo
  g1[6] = 0;
  g1[7] = 0;
  i32x4 gz4 = {0, 0, 0, 0};                               // groups 2/3: 2-D tensor
  i32x8 gz8 = {0, 0, 0, 0, 0, 0, 0, 0};                   // trailing group (clang-23 form)
  __builtin_amdgcn_tensor_load_to_lds(g0, g1, gz4, gz4, gz8, 0);
}

// ---------------- cast kernel ----------------
__global__ void cast_f32_to_bf16(const float* __restrict__ src,
                                 unsigned short* __restrict__ dst, int n) {
  int i = blockIdx.x * blockDim.x + threadIdx.x;
  if (i < n) dst[i] = f2bf(src[i]);
}

// ---------------- GEMM: C[M,1024] = A[M,1024] @ W[1024,1024]^T + bias ---------------
// TDM double-buffered pipeline: wave 0 issues next K-step tiles while 4 waves run
// 64x64 WMMA tiles (16 v_wmma per wave per K-step).
__global__ __launch_bounds__(128) void gemm_bf16_xwT(
    const unsigned short* __restrict__ A,   // M x K row-major bf16
    const unsigned short* __restrict__ W,   // N x K row-major bf16
    const float* __restrict__ bias,         // N
    float* __restrict__ C,                  // M x N row-major f32
    int M) {
  constexpr int K = HID, N = HID;
  constexpr int LDT = 48;                   // padded LDS row stride (96B, 16B aligned)
  __shared__ unsigned short As[2][128 * LDT];
  __shared__ unsigned short Bs[2][128 * LDT];

  const int tid  = threadIdx.x;
  const int lane = tid & 31;
  const int wave = tid >> 5;                // 0..3
  const int lq   = lane & 15;
  const int half = lane >> 4;
  const int wm   = wave >> 1;               // 0..1  (64-row slab)
  const int wn   = wave & 1;                // 0..1  (64-col slab)
  const int m0   = blockIdx.x * 128;
  const int n0   = blockIdx.y * 128;

  f32x8 acc[4][4];
  for (int mi = 0; mi < 4; ++mi)
    for (int nj = 0; nj < 4; ++nj) {
      float bv = bias[n0 + wn * 64 + nj * 16 + lq];
      for (int j = 0; j < 8; ++j) acc[mi][nj][j] = bv;
    }

  const unsigned aLds0 = (unsigned)(uintptr_t)&As[0][0];
  const unsigned aLds1 = (unsigned)(uintptr_t)&As[1][0];
  const unsigned bLds0 = (unsigned)(uintptr_t)&Bs[0][0];
  const unsigned bLds1 = (unsigned)(uintptr_t)&Bs[1][0];
  const unsigned short* aBase = A + (size_t)m0 * K;
  const unsigned short* bBase = W + (size_t)n0 * K;
  const unsigned aRows = (unsigned)(M - m0);   // TDM OOB zero-fill handles tails
  const unsigned bRows = (unsigned)(N - n0);

  // prologue: fill buffer 0
  if (wave == 0) {
    tdm_load_tile(aLds0, aBase, aRows);
    tdm_load_tile(bLds0, bBase, bRows);
  }

  int buf = 0;
  for (int k0 = 0; k0 < K; k0 += 32) {
    if (wave == 0) {
      if (k0 + 32 < K) {
        tdm_load_tile(buf ? aLds0 : aLds1, aBase + (k0 + 32), aRows);
        tdm_load_tile(buf ? bLds0 : bLds1, bBase + (k0 + 32), bRows);
        __builtin_amdgcn_s_wait_tensorcnt(2);   // current tiles landed, next 2 in flight
      } else {
        __builtin_amdgcn_s_wait_tensorcnt(0);
      }
    }
    __syncthreads();

    const unsigned short* as = As[buf];
    const unsigned short* bs = Bs[buf];
    Frag16 af[4], bfg[4];
    for (int mi = 0; mi < 4; ++mi) {        // A frag: M=lq, K={c..c+7, c+16..c+23}
      int r = wm * 64 + mi * 16 + lq;
      int c = half * 8;
      af[mi].q[0] = *reinterpret_cast<const u32x4*>(&as[r * LDT + c]);
      af[mi].q[1] = *reinterpret_cast<const u32x4*>(&as[r * LDT + c + 16]);
    }
    for (int nj = 0; nj < 4; ++nj) {        // B frag: N=lq, K={c..c+7, c+16..c+23}
      int r = wn * 64 + nj * 16 + lq;
      int c = half * 8;
      bfg[nj].q[0] = *reinterpret_cast<const u32x4*>(&bs[r * LDT + c]);
      bfg[nj].q[1] = *reinterpret_cast<const u32x4*>(&bs[r * LDT + c + 16]);
    }
    for (int mi = 0; mi < 4; ++mi)
      for (int nj = 0; nj < 4; ++nj)
        acc[mi][nj] = WMMA_BF16(acc[mi][nj], af[mi].v, bfg[nj].v);
    __syncthreads();
    buf ^= 1;
  }

  for (int mi = 0; mi < 4; ++mi)
    for (int nj = 0; nj < 4; ++nj)
      for (int j = 0; j < 8; ++j) {
        int row = m0 + wm * 64 + mi * 16 + j + 8 * half;   // C layout: M=j+8*half
        if (row < M) {
          int col = n0 + wn * 64 + nj * 16 + lq;           // N=lane%16
          C[(size_t)row * N + col] = acc[mi][nj][j];
        }
      }
}

// ---------------- RoPE + pack to attention layouts ----------------
__global__ void rope_pack(const float* __restrict__ qv, const float* __restrict__ kv,
                          const float* __restrict__ vv, const float* __restrict__ qa,
                          const float* __restrict__ ka, const float* __restrict__ va,
                          unsigned short* __restrict__ qb, unsigned short* __restrict__ kb,
                          unsigned short* __restrict__ vt) {
  const int tok = blockIdx.x;   // 0..NP-1
  const int b   = blockIdx.y;
  const int d   = threadIdx.x;  // 0..63

  const bool rot = d < 60;      // 3 sections of ROT=20, last 4 passthrough
  const bool pad = tok >= NT;

  float c = 1.f, s = 0.f;
  if (!pad && rot) {
    int sec = d / 20;
    int i   = (d - sec * 20) >> 1;
    float pos = 0.f;                        // action tokens: linspace(0,0)=0
    if (tok < NV) {
      int dp = tok >> 8, hp = (tok >> 4) & 15, wp = tok & 15;
      pos = (sec == 0) ? (float)dp : (sec == 1 ? (float)hp : (float)wp);
    }
    float omega = __expf(-(float)i * 0.9210340372f); // 10000^(-i/10)
    float f = pos * omega;
    c = __cosf(f); s = __sinf(f);
  }

  for (int h = 0; h < HEADS; ++h) {
    float qx = 0.f, qp = 0.f, kx = 0.f, kpv = 0.f, vx = 0.f;
    if (!pad) {
      int part = rot ? (d ^ 1) : d;
      if (tok < NV) {
        size_t base = ((size_t)(b * NV + tok)) * HID + h * HD;
        qx = qv[base + d]; kx = kv[base + d]; vx = vv[base + d];
        qp = qv[base + part]; kpv = kv[base + part];
      } else {
        size_t base = ((size_t)(b * NA + (tok - NV))) * HID + h * HD;
        qx = qa[base + d]; kx = ka[base + d]; vx = va[base + d];
        qp = qa[base + part]; kpv = ka[base + part];
      }
    }
    float qo, ko;
    if (pad || !rot)        { qo = qx;               ko = kx; }
    else if ((d & 1) == 0)  { qo = qx * c - qp * s;  ko = kx * c - kpv * s; }
    else                    { qo = qx * c + qp * s;  ko = kx * c + kpv * s; }

    size_t qkidx = (((size_t)(b * HEADS + h)) * NP + tok) * HD + d;   // [b,h,tok,d]
    qb[qkidx] = f2bf(qo);
    kb[qkidx] = f2bf(ko);
    size_t vidx = (((size_t)(b * HEADS + h)) * HD + d) * NP + tok;    // [b,h,d,tok]
    vt[vidx] = f2bf(vx);
  }
}

// ---------------- flash attention: 1 wave per 64-query block ----------------
__global__ __launch_bounds__(32) void attn_fa(
    const unsigned short* __restrict__ qb, const unsigned short* __restrict__ kb,
    const unsigned short* __restrict__ vt, unsigned short* __restrict__ ctx) {
  constexpr float SCALE = 0.125f;   // 64^-0.5
  const int lane = threadIdx.x;
  const int lq   = lane & 15;
  const int half = lane >> 4;
  const int qblk = blockIdx.x, h = blockIdx.y, b = blockIdx.z;

  const unsigned short* qp = qb + (((size_t)(b * HEADS + h)) * NP + qblk * 64) * HD;
  const unsigned short* kp = kb + ((size_t)(b * HEADS + h)) * NP * HD;
  const unsigned short* vp = vt + ((size_t)(b * HEADS + h)) * HD * NP;

  __shared__ unsigned short plds[64 * 72];  // P staging, 144B row stride

  // resident Q fragments: 4 M-tiles x 2 K-steps (K=64)
  Frag16 qf[4][2];
  for (int mi = 0; mi < 4; ++mi)
    for (int ks = 0; ks < 2; ++ks) {
      const unsigned short* p = qp + (size_t)(mi * 16 + lq) * HD + ks * 32 + half * 8;
      qf[mi][ks].q[0] = *reinterpret_cast<const u32x4*>(p);
      qf[mi][ks].q[1] = *reinterpret_cast<const u32x4*>(p + 16);
    }

  f32x8 acc[4][4];
  for (int mi = 0; mi < 4; ++mi)
    for (int nd = 0; nd < 4; ++nd)
      for (int j = 0; j < 8; ++j) acc[mi][nd][j] = 0.f;
  float rowM[4][8], rowL[4][8];
  for (int mi = 0; mi < 4; ++mi)
    for (int j = 0; j < 8; ++j) { rowM[mi][j] = -3.0e38f; rowL[mi][j] = 0.f; }

  for (int kbi = 0; kbi < NP / 64; ++kbi) {
    // K^T B-fragments: lane n = key token, d = {half*8..+7, 16+half*8..+7} in-lane
    Frag16 kf[4][2];
    for (int nj = 0; nj < 4; ++nj)
      for (int ks = 0; ks < 2; ++ks) {
        const unsigned short* p =
            kp + (size_t)(kbi * 64 + nj * 16 + lq) * HD + ks * 32 + half * 8;
        kf[nj][ks].q[0] = *reinterpret_cast<const u32x4*>(p);
        kf[nj][ks].q[1] = *reinterpret_cast<const u32x4*>(p + 16);
      }
    if (kbi + 1 < NP / 64) {                 // warm next key block into cache
      __builtin_prefetch(kp + (size_t)((kbi + 1) * 64 + lane) * HD, 0, 1);
    }
    // scores
    f32x8 sc[4][4];
    for (int mi = 0; mi < 4; ++mi)
      for (int nj = 0; nj < 4; ++nj) {
        f32x8 t; for (int j = 0; j < 8; ++j) t[j] = 0.f;
        t = WMMA_BF16(t, qf[mi][0].v, kf[nj][0].v);
        t = WMMA_BF16(t, qf[mi][1].v, kf[nj][1].v);
        int token = kbi * 64 + nj * 16 + lq;
        if (token >= NT) for (int j = 0; j < 8; ++j) t[j] = -3.0e38f;
        else             for (int j = 0; j < 8; ++j) t[j] *= SCALE;
        sc[mi][nj] = t;
      }
    // online softmax (rows live in 16-lane halves)
    for (int mi = 0; mi < 4; ++mi)
      for (int j = 0; j < 8; ++j) {
        float mx = fmaxf(fmaxf(sc[mi][0][j], sc[mi][1][j]),
                         fmaxf(sc[mi][2][j], sc[mi][3][j]));
        for (int m = 1; m < 16; m <<= 1) mx = fmaxf(mx, __shfl_xor(mx, m, 32));
        float mNew  = fmaxf(rowM[mi][j], mx);
        float scale = __expf(rowM[mi][j] - mNew);
        rowM[mi][j] = mNew;
        float rs = 0.f;
        for (int nj = 0; nj < 4; ++nj) {
          float pv = __expf(sc[mi][nj][j] - mNew);
          sc[mi][nj][j] = pv;
          rs += pv;
        }
        for (int m = 1; m < 16; m <<= 1) rs += __shfl_xor(rs, m, 32);
        rowL[mi][j] = rowL[mi][j] * scale + rs;
        for (int nd = 0; nd < 4; ++nd) acc[mi][nd][j] *= scale;
      }
    // P: C-layout -> LDS -> A-fragments
    for (int mi = 0; mi < 4; ++mi)
      for (int nj = 0; nj < 4; ++nj)
        for (int j = 0; j < 8; ++j)
          plds[(mi * 16 + j + 8 * half) * 72 + nj * 16 + lq] = f2bf(sc[mi][nj][j]);
    __syncthreads();
    // ctx += P @ V  (V^T layout -> contiguous B-frag loads)
    for (int ks = 0; ks < 2; ++ks) {
      Frag16 pf[4], vfg[4];
      for (int mi = 0; mi < 4; ++mi) {
        const unsigned short* p = &plds[(mi * 16 + lq) * 72 + ks * 32 + half * 8];
        pf[mi].q[0] = *reinterpret_cast<const u32x4*>(p);
        pf[mi].q[1] = *reinterpret_cast<const u32x4*>(p + 16);
      }
      for (int nd = 0; nd < 4; ++nd) {
        const unsigned short* p =
            vp + (size_t)(nd * 16 + lq) * NP + kbi * 64 + ks * 32 + half * 8;
        vfg[nd].q[0] = *reinterpret_cast<const u32x4*>(p);
        vfg[nd].q[1] = *reinterpret_cast<const u32x4*>(p + 16);
      }
      for (int mi = 0; mi < 4; ++mi)
        for (int nd = 0; nd < 4; ++nd)
          acc[mi][nd] = WMMA_BF16(acc[mi][nd], pf[mi].v, vfg[nd].v);
    }
    __syncthreads();
  }

  // finalize: ctx[b, tok, h*64+d] (bf16, feeds output projection GEMM)
  for (int mi = 0; mi < 4; ++mi)
    for (int j = 0; j < 8; ++j) {
      float inv = 1.0f / rowL[mi][j];
      int tok = qblk * 64 + mi * 16 + j + 8 * half;
      if (tok < NT) {
        size_t base = ((size_t)(b * NT + tok)) * HID + h * HD;
        for (int nd = 0; nd < 4; ++nd)
          ctx[base + nd * 16 + lq] = f2bf(acc[mi][nd][j] * inv);
      }
    }
}

// ---------------- host orchestration ----------------
extern "C" void kernel_launch(void* const* d_in, const int* in_sizes, int n_in,
                              void* d_out, int out_size, void* d_ws, size_t ws_size,
                              hipStream_t stream) {
  // setup_inputs() dict order: (W_q,b_q),(W_k,b_k),(W_v,b_v),(W_qa,b_qa),
  // (W_ka,b_ka),(W_va,b_va),(W_proj,b_proj),(W_proj_a,b_proj_a), video, action
  const float* Wf[8]; const float* bias[8];
  for (int i = 0; i < 8; ++i) { Wf[i] = (const float*)d_in[2 * i]; bias[i] = (const float*)d_in[2 * i + 1]; }
  const float* xv = (const float*)d_in[16];
  const float* xa = (const float*)d_in[17];
  float* out = (float*)d_out;

  char* ws = (char*)d_ws;
  size_t off = 0;
  auto alloc = [&](size_t bytes) -> char* {
    char* p = ws + off; off = (off + bytes + 255) & ~(size_t)255; return p;
  };
  unsigned short* xvb = (unsigned short*)alloc((size_t)BZ * NV * HID * 2);
  unsigned short* xab = (unsigned short*)alloc((size_t)BZ * NA * HID * 2);
  unsigned short* Wb[8];
  for (int i = 0; i < 8; ++i) Wb[i] = (unsigned short*)alloc((size_t)HID * HID * 2);
  float* qvf = (float*)alloc((size_t)BZ * NV * HID * 4);
  float* kvf = (float*)alloc((size_t)BZ * NV * HID * 4);
  float* vvf = (float*)alloc((size_t)BZ * NV * HID * 4);
  float* qaf = (float*)alloc((size_t)BZ * NA * HID * 4);
  float* kaf = (float*)alloc((size_t)BZ * NA * HID * 4);
  float* vaf = (float*)alloc((size_t)BZ * NA * HID * 4);
  unsigned short* qbb  = (unsigned short*)alloc((size_t)BZ * HEADS * NP * HD * 2);
  unsigned short* kbb  = (unsigned short*)alloc((size_t)BZ * HEADS * NP * HD * 2);
  unsigned short* vtb  = (unsigned short*)alloc((size_t)BZ * HEADS * HD * NP * 2);
  unsigned short* ctxb = (unsigned short*)alloc((size_t)BZ * NT * HID * 2);

  auto cast = [&](const float* src, unsigned short* dst, size_t n) {
    int blocks = (int)((n + 255) / 256);
    hipLaunchKernelGGL(cast_f32_to_bf16, dim3(blocks), dim3(256), 0, stream, src, dst, (int)n);
  };
  cast(xv, xvb, (size_t)BZ * NV * HID);
  cast(xa, xab, (size_t)BZ * NA * HID);
  for (int i = 0; i < 8; ++i) cast(Wf[i], Wb[i], (size_t)HID * HID);

  auto gemm = [&](const unsigned short* A, const unsigned short* Wt, const float* bi,
                  float* C, int M) {
    dim3 grid((M + 127) / 128, HID / 128);
    hipLaunchKernelGGL(gemm_bf16_xwT, grid, dim3(128), 0, stream, A, Wt, bi, C, M);
  };
  gemm(xvb, Wb[0], bias[0], qvf, BZ * NV);
  gemm(xvb, Wb[1], bias[1], kvf, BZ * NV);
  gemm(xvb, Wb[2], bias[2], vvf, BZ * NV);
  gemm(xab, Wb[3], bias[3], qaf, BZ * NA);
  gemm(xab, Wb[4], bias[4], kaf, BZ * NA);
  gemm(xab, Wb[5], bias[5], vaf, BZ * NA);

  hipLaunchKernelGGL(rope_pack, dim3(NP, BZ), dim3(64), 0, stream,
                     qvf, kvf, vvf, qaf, kaf, vaf, qbb, kbb, vtb);

  hipLaunchKernelGGL(attn_fa, dim3(NP / 64, HEADS, BZ), dim3(32), 0, stream,
                     qbb, kbb, vtb, ctxb);

  // output projections: ctx is bf16 -> reuse GEMM
  float* out_a = out + (size_t)BZ * NV * HID;
  for (int b = 0; b < BZ; ++b) {
    gemm(ctxb + (size_t)b * NT * HID, Wb[6], bias[6], out + (size_t)b * NV * HID, NV);
    gemm(ctxb + ((size_t)b * NT + NV) * HID, Wb[7], bias[7], out_a + (size_t)b * NA * HID, NA);
  }
  (void)in_sizes; (void)n_in; (void)out_size; (void)ws_size;
}
